// LocalPoolPointnet_56014963474825
// MI455X (gfx1250) — compile-verified, matrix operand order checked by reading, not run on hardware
//
#include <hip/hip_runtime.h>
#include <hip/hip_bf16.h>

// ---------------- CDNA5 WMMA types ----------------
typedef __attribute__((ext_vector_type(16))) __bf16 v16bf;
typedef __attribute__((ext_vector_type(8)))  float  v8f;
typedef __attribute__((ext_vector_type(8)))  unsigned short v8us;

#define NPTS   160000   // B*T
#define TPTS   20000    // T (points per batch)
#define NB     8        // B
#define HID    128
#define SRES   4096     // 64*64
#define RESO   64

// ---------------- helpers ----------------
__device__ __forceinline__ unsigned short f2us(float f) {
    // float -> bf16 bits, round-to-nearest-even
    unsigned u = __float_as_uint(f);
    unsigned r = u + 0x7FFFu + ((u >> 16) & 1u);
    return (unsigned short)(r >> 16);
}

__device__ __forceinline__ int grid_i(float v) {
    float u = v * (1.0f / 1.101f) + 0.5f;          // /(1+PADDING+1e-3)
    u = fminf(fmaxf(u, 0.0f), 1.0f - 1e-3f);
    return (int)(u * (float)RESO);
}

__device__ __forceinline__ void atomic_fmax(float* addr, float val) {
#if defined(__has_builtin) && __has_builtin(__hip_atomic_fetch_max)
    __hip_atomic_fetch_max(addr, val, __ATOMIC_RELAXED, __HIP_MEMORY_SCOPE_AGENT);
#else
    unsigned* ua = (unsigned*)addr;
    unsigned old = *ua;
    while (true) {
        float cur = __uint_as_float(old);
        float nw  = fmaxf(cur, val);               // fmaxf(NaN,x)=x -> NaN init works
        unsigned nu = __float_as_uint(nw);
        if (nu == old) break;
        unsigned prev = atomicCAS(ua, old, nu);
        if (prev == old) break;
        old = prev;
    }
#endif
}

// Load a 16x32 bf16 A-fragment from a row-major LDS tile.
// Lane L holds row m=L&15; K layout per CDNA5 ISA: element e ->
// k = (e/8)*16 + (L>>4)*8 + (e%8), i.e. two contiguous 8-element runs.
__device__ __forceinline__ v16bf load_a_frag(const unsigned short* rowbase, int kt, int hl, bool doRelu) {
    const v8us lo = *(const v8us*)(rowbase + kt * 32 + hl * 8);
    const v8us hi = *(const v8us*)(rowbase + kt * 32 + 16 + hl * 8);
    union { v16bf v; v8us u[2]; unsigned short s[16]; } r;
    r.u[0] = lo; r.u[1] = hi;
    if (doRelu) {
#pragma unroll
        for (int i = 0; i < 16; ++i)
            if (r.s[i] & 0x8000u) r.s[i] = 0;      // bf16 relu: negative -> 0
    }
    return r.v;
}

// ---------------- weight pre-swizzle ----------------
// dst[((kt*8+nt)*32+lane)*16+e] = bf16(W[k*128+n]) with the ISA 16-bit B layout
__global__ void pack_w(const float* __restrict__ src, unsigned short* __restrict__ dst, int KT) {
    int t = blockIdx.x * blockDim.x + threadIdx.x;
    int total = KT * 8 * 512;
    if (t >= total) return;
    int e    = t & 15;
    int lane = (t >> 4) & 31;
    int fr   = t >> 9;
    int nt   = fr & 7;
    int kt   = fr >> 3;
    int hl   = lane >> 4;
    int n    = nt * 16 + (lane & 15);
    int k    = kt * 32 + (e >> 3) * 16 + hl * 8 + (e & 7);
    dst[t] = f2us(src[k * 128 + n]);
}

// ---------------- pipelined dual-M-tile GEMM ----------------
// Two 16-row A tiles share each B fragment (2 WMMAs per 1KB of weights),
// and the next B fragment is kept in flight while the current pair executes.
template<int KT, bool RELU>
__device__ __forceinline__ void gemm2(
    const unsigned short* row0, const unsigned short* row1,
    const unsigned short* wp, int lane, int hl,
    v8f accA[8], v8f accB[8])
{
    const v16bf* bp = (const v16bf*)wp + lane;
    v16bf bc = bp[0];
#pragma unroll
    for (int kt = 0; kt < KT; ++kt) {
        v16bf a0 = load_a_frag(row0, kt, hl, RELU);
        v16bf a1 = load_a_frag(row1, kt, hl, RELU);
#pragma unroll
        for (int nt = 0; nt < 8; ++nt) {
            const int lin = kt * 8 + nt;
            v16bf bn = (lin + 1 < KT * 8) ? bp[(size_t)(lin + 1) * 32] : bc;
            accA[nt] = __builtin_amdgcn_wmma_f32_16x16x32_bf16(false, a0, false, bc,
                                                               (short)0, accA[nt], false, false);
            accB[nt] = __builtin_amdgcn_wmma_f32_16x16x32_bf16(false, a1, false, bc,
                                                               (short)0, accB[nt], false, false);
            bc = bn;
        }
    }
}

// ---------------- ResNet block on one wave (32 pts x 128 out) ----------------
__device__ __forceinline__ void resblock_wave(
    const unsigned short* xb, unsigned short* hb,
    const unsigned short* w0p, const unsigned short* w1p, const unsigned short* wsp,
    const float* b0, const float* b1,
    float* __restrict__ net, long ptbase)
{
    const int lane  = threadIdx.x & 31;
    const int hl    = lane >> 4;
    const int col16 = lane & 15;
    const unsigned short* xrow0 = xb + (lane & 15) * 256;
    const unsigned short* xrow1 = xb + (16 + (lane & 15)) * 256;

    __builtin_prefetch(w0p, 0, 0);   // global_prefetch_b8 on the weight stream

    // ---- h = relu(x) @ w0 ----
    v8f accA[8] = {}, accB[8] = {};
    gemm2<8, true>(xrow0, xrow1, w0p, lane, hl, accA, accB);

    // bias + relu, stash h in LDS (bf16)
#pragma unroll
    for (int nt = 0; nt < 8; ++nt) {
        float bias = b0[nt * 16 + col16];
#pragma unroll
        for (int r = 0; r < 8; ++r) {
            float vA = accA[nt][r] + bias; vA = vA > 0.0f ? vA : 0.0f;
            float vB = accB[nt][r] + bias; vB = vB > 0.0f ? vB : 0.0f;
            hb[(r + 8 * hl) * 128 + nt * 16 + col16]      = f2us(vA);
            hb[(16 + r + 8 * hl) * 128 + nt * 16 + col16] = f2us(vB);
        }
    }

    // ---- out = h @ w1 + x @ ws (+ b1) ----
    v8f oA[8] = {}, oB[8] = {};
    const unsigned short* hrow0 = hb + (lane & 15) * 128;
    const unsigned short* hrow1 = hb + (16 + (lane & 15)) * 128;
    gemm2<4, false>(hrow0, hrow1, w1p, lane, hl, oA, oB);
    gemm2<8, false>(xrow0, xrow1, wsp, lane, hl, oA, oB);   // shortcut uses raw x

    // store net (fp32)
#pragma unroll
    for (int nt = 0; nt < 8; ++nt) {
        float bias = b1[nt * 16 + col16];
#pragma unroll
        for (int r = 0; r < 8; ++r) {
            long ptA = ptbase + r + 8 * hl;
            long ptB = ptA + 16;
            net[ptA * 128 + nt * 16 + col16] = oA[nt][r] + bias;
            net[ptB * 128 + nt * 16 + col16] = oB[nt][r] + bias;
        }
    }
}

// ---------------- kernel 1: indices + fc_pos + block0 ----------------
__global__ void __launch_bounds__(64) k_init(
    const float* __restrict__ p, const float* __restrict__ fcw, const float* __restrict__ fcb,
    const unsigned short* __restrict__ w0p, const unsigned short* __restrict__ w1p,
    const unsigned short* __restrict__ wsp,
    const float* __restrict__ b0, const float* __restrict__ b1,
    int* __restrict__ idxbuf, float* __restrict__ net)
{
    __shared__ __align__(16) unsigned short xb_s[2][32 * 256];
    __shared__ __align__(16) unsigned short hb_s[2][32 * 128];
    const int wid = threadIdx.x >> 5, lane = threadIdx.x & 31;
    unsigned short* xb = xb_s[wid];
    unsigned short* hb = hb_s[wid];
    const long ptbase = (long)(blockIdx.x * 2 + wid) * 32;
    const long pt = ptbase + lane;

    const float p0 = p[pt * 3 + 0], p1 = p[pt * 3 + 1], p2 = p[pt * 3 + 2];

    // fc_pos: each lane fills its point's full 256-wide row
#pragma unroll 4
    for (int j = 0; j < 256; ++j) {
        float v = fcb[j] + p0 * fcw[j] + p1 * fcw[256 + j] + p2 * fcw[512 + j];
        xb[lane * 256 + j] = f2us(v);
    }
    // plane indices: (0,2)=xz, (0,1)=xy, (1,2)=yz
    idxbuf[0 * NPTS + pt] = grid_i(p0) + RESO * grid_i(p2);
    idxbuf[1 * NPTS + pt] = grid_i(p0) + RESO * grid_i(p1);
    idxbuf[2 * NPTS + pt] = grid_i(p1) + RESO * grid_i(p2);

    resblock_wave(xb, hb, w0p, w1p, wsp, b0, b1, net, ptbase);
}

// ---------------- kernel 2: scatter-max into plane grids ----------------
__global__ void __launch_bounds__(256) k_scatter_max(
    const float* __restrict__ net, const int* __restrict__ idxbuf, float* __restrict__ grids)
{
    long t = (long)blockIdx.x * 256 + threadIdx.x;     // NPTS*32 threads
    long pt = t >> 5;
    if (pt >= NPTS) return;
    int fq = (int)(t & 31) * 4;
    int b = (int)(pt / TPTS);
    const float4 v = *(const float4*)(net + pt * 128 + fq);
#pragma unroll
    for (int pl = 0; pl < 3; ++pl) {
        int id = idxbuf[pl * NPTS + pt];
        float* g = grids + (((long)(pl * NB + b) * SRES + id) * 128 + fq);
        atomic_fmax(g + 0, v.x);
        atomic_fmax(g + 1, v.y);
        atomic_fmax(g + 2, v.z);
        atomic_fmax(g + 3, v.w);
    }
}

// ---------------- kernel 3: gather + concat + resblock ----------------
__global__ void __launch_bounds__(64) k_round(
    float* __restrict__ net, const float* __restrict__ grids, const int* __restrict__ idxbuf,
    const unsigned short* __restrict__ w0p, const unsigned short* __restrict__ w1p,
    const unsigned short* __restrict__ wsp,
    const float* __restrict__ b0, const float* __restrict__ b1)
{
    __shared__ __align__(16) unsigned short xb_s[2][32 * 256];
    __shared__ __align__(16) unsigned short hb_s[2][32 * 128];
    const int wid = threadIdx.x >> 5, lane = threadIdx.x & 31;
    unsigned short* xb = xb_s[wid];
    unsigned short* hb = hb_s[wid];
    const long ptbase = (long)(blockIdx.x * 2 + wid) * 32;

    // build x = [net | sum_planes gather(max-grid)]
    for (int e = lane; e < 32 * 128; e += 32) {
        int mm = e >> 7, f = e & 127;
        long pt = ptbase + mm;
        int b = (int)(pt / TPTS);
        xb[mm * 256 + f] = f2us(net[pt * 128 + f]);
        float s = 0.0f;
#pragma unroll
        for (int pl = 0; pl < 3; ++pl) {
            int id = idxbuf[pl * NPTS + pt];
            s += grids[((long)(pl * NB + b) * SRES + id) * 128 + f];
        }
        xb[mm * 256 + 128 + f] = f2us(s);
    }
    resblock_wave(xb, hb, w0p, w1p, wsp, b0, b1, net, ptbase);
}

// ---------------- kernel 4: fc_c + scatter-add (mean numerator) ----------------
__global__ void __launch_bounds__(64) k_final(
    const float* __restrict__ net, const int* __restrict__ idxbuf,
    const unsigned short* __restrict__ wcp, const float* __restrict__ cb,
    float* __restrict__ out, float* __restrict__ cnt)
{
    __shared__ __align__(16) unsigned short xb_s[2][32 * 128];
    const int wid = threadIdx.x >> 5, lane = threadIdx.x & 31;
    const int hl = lane >> 4, col16 = lane & 15;
    unsigned short* xb = xb_s[wid];
    const long ptbase = (long)(blockIdx.x * 2 + wid) * 32;

    for (int e = lane; e < 32 * 128; e += 32) {
        int mm = e >> 7, f = e & 127;
        xb[mm * 128 + f] = f2us(net[(ptbase + mm) * 128 + f]);
    }

    v8f accA[8] = {}, accB[8] = {};
    const unsigned short* xrow0 = xb + (lane & 15) * 128;
    const unsigned short* xrow1 = xb + (16 + (lane & 15)) * 128;
    gemm2<4, false>(xrow0, xrow1, wcp, lane, hl, accA, accB);

    // scatter-add c into out[3][B][C][S]
#pragma unroll
    for (int nt = 0; nt < 8; ++nt) {
        float bias = cb[nt * 16 + col16];
        int f = nt * 16 + col16;
#pragma unroll
        for (int r = 0; r < 8; ++r) {
#pragma unroll
            for (int half = 0; half < 2; ++half) {
                long pt = ptbase + 16 * half + r + 8 * hl;
                int b = (int)(pt / TPTS);
                float v = (half ? accB[nt][r] : accA[nt][r]) + bias;
#pragma unroll
                for (int pl = 0; pl < 3; ++pl) {
                    int id = idxbuf[pl * NPTS + pt];
                    atomicAdd(out + (((long)(pl * NB + b) * 128 + f) * SRES + id), v);
                }
            }
        }
    }
    // bin counts (one per point per plane)
    {
        long pt = ptbase + lane;
        int b = (int)(pt / TPTS);
#pragma unroll
        for (int pl = 0; pl < 3; ++pl)
            atomicAdd(cnt + ((long)(pl * NB + b) * SRES + idxbuf[pl * NPTS + pt]), 1.0f);
    }
}

// ---------------- kernel 5: mean normalize ----------------
__global__ void __launch_bounds__(256) k_div(float* __restrict__ out, const float* __restrict__ cnt) {
    long t = (long)blockIdx.x * 256 + threadIdx.x;   // 3*B*128*4096 elements
    if (t >= (long)3 * NB * 128 * SRES) return;
    long s  = t & (SRES - 1);
    long pb = (t >> 12) >> 7;                         // pl*NB + b
    float c = cnt[pb * SRES + s];
    out[t] = out[t] / fmaxf(c, 1.0f);
}

// ---------------- host launcher ----------------
extern "C" void kernel_launch(void* const* d_in, const int* in_sizes, int n_in,
                              void* d_out, int out_size, void* d_ws, size_t ws_size,
                              hipStream_t stream) {
    const float* p        = (const float*)d_in[0];
    const float* fc_pos_w = (const float*)d_in[1];
    const float* fc_pos_b = (const float*)d_in[2];
    const float* blk0_w   = (const float*)d_in[3];   // [5][256][128]
    const float* blk0_b   = (const float*)d_in[4];   // [5][128]
    const float* blk1_w   = (const float*)d_in[5];   // [5][128][128]
    const float* blk1_b   = (const float*)d_in[6];   // [5][128]
    const float* blk_sc_w = (const float*)d_in[7];   // [5][256][128]
    const float* fc_c_w   = (const float*)d_in[8];   // [128][128]
    const float* fc_c_b   = (const float*)d_in[9];   // [128]
    float* out = (float*)d_out;

    // workspace carve
    char* ws = (char*)d_ws;
    int*   idxbuf = (int*)ws;                 ws += (size_t)3 * NPTS * sizeof(int);
    float* net    = (float*)ws;               ws += (size_t)NPTS * 128 * sizeof(float);
    float* grids  = (float*)ws;               ws += (size_t)3 * NB * SRES * 128 * sizeof(float);
    float* cnt    = (float*)ws;               ws += (size_t)3 * NB * SRES * sizeof(float);
    unsigned short* wpk = (unsigned short*)ws;
    const size_t SZ_W0 = 8 * 8 * 512;   // 32768 bf16 per 256x128 matrix
    const size_t SZ_W1 = 4 * 8 * 512;   // 16384 bf16 per 128x128 matrix
    unsigned short* W0P = wpk;
    unsigned short* W1P = W0P + 5 * SZ_W0;
    unsigned short* WSP = W1P + 5 * SZ_W1;
    unsigned short* WCP = WSP + 5 * SZ_W0;

    // 1) pre-swizzle weights into WMMA fragment layout (bf16)
    for (int i = 0; i < 5; ++i) {
        pack_w<<<(int)(SZ_W0 + 255) / 256, 256, 0, stream>>>(blk0_w   + (size_t)i * 256 * 128, W0P + i * SZ_W0, 8);
        pack_w<<<(int)(SZ_W1 + 255) / 256, 256, 0, stream>>>(blk1_w   + (size_t)i * 128 * 128, W1P + i * SZ_W1, 4);
        pack_w<<<(int)(SZ_W0 + 255) / 256, 256, 0, stream>>>(blk_sc_w + (size_t)i * 256 * 128, WSP + i * SZ_W0, 8);
    }
    pack_w<<<(int)(SZ_W1 + 255) / 256, 256, 0, stream>>>(fc_c_w, WCP, 4);

    const int GB = NPTS / 64;   // 2500 blocks, 2 waves x 32 pts each

    // 2) indices + fc_pos + block 0
    k_init<<<GB, 64, 0, stream>>>(p, fc_pos_w, fc_pos_b,
                                  W0P, W1P, WSP, blk0_b, blk1_b, idxbuf, net);

    // 3) rounds 1..4: scatter-max pool -> gather -> resblock
    const size_t grid_bytes = (size_t)3 * NB * SRES * 128 * sizeof(float);
    for (int i = 1; i < 5; ++i) {
        hipMemsetAsync(grids, 0xFF, grid_bytes, stream);   // NaN init; max_num(NaN,x)=x
        k_scatter_max<<<NPTS * 32 / 256, 256, 0, stream>>>(net, idxbuf, grids);
        k_round<<<GB, 64, 0, stream>>>(net, grids, idxbuf,
                                       W0P + i * SZ_W0, W1P + i * SZ_W1, WSP + i * SZ_W0,
                                       blk0_b + i * 128, blk1_b + i * 128);
    }

    // 4) fc_c + scatter-mean into output planes
    hipMemsetAsync(out, 0, (size_t)out_size * sizeof(float), stream);
    hipMemsetAsync(cnt, 0, (size_t)3 * NB * SRES * sizeof(float), stream);
    k_final<<<GB, 64, 0, stream>>>(net, idxbuf, WCP, fc_c_b, out, cnt);

    const long nout = (long)3 * NB * 128 * SRES;
    k_div<<<(int)((nout + 255) / 256), 256, 0, stream>>>(out, cnt);
}
